// SAFrame_52656299049418
// MI455X (gfx1250) — compile-verified, matrix operand order checked by reading, not run on hardware
//
#include <hip/hip_runtime.h>

// ---------------------------------------------------------------------------
// SAFrame hetero-GNN on MI455X (gfx1250, wave32, WMMA)
//
// Pass 1: transpose/convert 7 agg W[256,128] and 3 sem Ws[128,128] to bf16,
//         laid out column-major so WMMA B-fragments are contiguous 16B loads.
// Pass 2: hete_tile_kernel — 16 rows/block, 128 threads (4 waves):
//         gather+mean into LDS (bf16, WMMA A layout compatible),
//         C[16,128] = X[16,256] @ W via v_wmma_f32_16x16x32_bf16 (f32 acc),
//         +bias, ReLU into LDS z[3][16][128]; then scoring GEMM
//         tanh(z@Ws+bs)@q via WMMA + shfl_xor reduce + atomics.
// Pass 3: softmax over the 3 global score means -> beta.
// Pass 4: out = sum_m beta[m]*z[:,m,:]  (pure bandwidth).
// Workspace: z_i(78.6MB)+z_l+z_t(3.1MB)+accums+bf16 weights (~82.4MB total).
// ---------------------------------------------------------------------------

#define DIM 128
#define KNB 20

typedef __bf16 v16bf __attribute__((ext_vector_type(16)));
typedef __bf16 v8bf  __attribute__((ext_vector_type(8)));
typedef __bf16 v4bf  __attribute__((ext_vector_type(4)));
typedef float  v8f   __attribute__((ext_vector_type(8)));

__device__ __forceinline__ v16bf mk16(v8bf lo, v8bf hi) {
  v16bf r;
#pragma unroll
  for (int i = 0; i < 8; ++i) { r[i] = lo[i]; r[i + 8] = hi[i]; }
  return r;
}

__device__ __forceinline__ v8bf cvt8(const float* p) {
  float4 f0 = *(const float4*)p;
  float4 f1 = *(const float4*)(p + 4);
  v8bf r;
  r[0] = (__bf16)f0.x; r[1] = (__bf16)f0.y; r[2] = (__bf16)f0.z; r[3] = (__bf16)f0.w;
  r[4] = (__bf16)f1.x; r[5] = (__bf16)f1.y; r[6] = (__bf16)f1.z; r[7] = (__bf16)f1.w;
  return r;
}

// Wt[n][k] = (bf16) W[k][n];  W is [Kdim][Ncol] row-major.
__global__ void transpose_bf16_kernel(const float* __restrict__ W,
                                      __bf16* __restrict__ Wt,
                                      int Kdim, int Ncol) {
  int j = blockIdx.x * blockDim.x + threadIdx.x;
  if (j < Kdim * Ncol) {
    int n = j / Kdim;
    int k = j % Kdim;
    Wt[j] = (__bf16)W[k * Ncol + n];
  }
}

__launch_bounds__(128)
__global__ void hete_tile_kernel(
    const int* __restrict__ idx, int use_session,
    const float* __restrict__ self_tab,
    const int* __restrict__ nb0, const float* __restrict__ tab0,
    const float* __restrict__ bias0, const __bf16* __restrict__ Wt0,
    const int* __restrict__ nb1, const float* __restrict__ tab1,
    const float* __restrict__ bias1, const __bf16* __restrict__ Wt1,
    const int* __restrict__ nb2, const float* __restrict__ tab2,
    const float* __restrict__ bias2, const __bf16* __restrict__ Wt2,
    const float* __restrict__ session,
    const __bf16* __restrict__ Wst, const float* __restrict__ bs,
    const float* __restrict__ q,
    float* __restrict__ z_out, float* __restrict__ wsum) {
  __shared__ int s_idx[16];
  __shared__ int s_nb[16][KNB];
  __shared__ __attribute__((aligned(16))) __bf16 s_X[16][272];   // [16][256]+pad
  __shared__ __attribute__((aligned(16))) float  s_z[3][16][DIM];
  __shared__ float s_w[3][16];

  const int tid  = threadIdx.x;          // 0..127
  const int lane = tid & 31;
  const int wave = tid >> 5;
  const int hf   = lane >> 4;            // 0 or 1 (wmma half)
  const int l15  = lane & 15;
  const int base = blockIdx.x * 16;

  if (tid < 16) s_idx[tid] = idx[base + tid];
  if (tid < 48) s_w[tid >> 4][tid & 15] = 0.0f;
  __syncthreads();

  // ---- self embedding -> X[:, 0:128] (bf16) -------------------------------
#pragma unroll
  for (int j = 0; j < 4; ++j) {
    int task = tid + j * 128;            // 16 rows x 32 float4
    int r = task >> 5, c4 = task & 31;
    const float4 v = *(const float4*)&self_tab[(long)s_idx[r] * DIM + c4 * 4];
    v4bf p; p[0] = (__bf16)v.x; p[1] = (__bf16)v.y; p[2] = (__bf16)v.z; p[3] = (__bf16)v.w;
    *(v4bf*)&s_X[r][c4 * 4] = p;
  }

  const int*    nbs[3]    = {nb0, nb1, nb2};
  const float*  tabs[3]   = {tab0, tab1, tab2};
  const float*  biases[3] = {bias0, bias1, bias2};
  const __bf16* Wts[3]    = {Wt0, Wt1, Wt2};

  const int n0a = wave * 32, n0b = n0a + 16;
  const int cola = n0a + l15, colb = n0b + l15;

#pragma unroll
  for (int m = 0; m < 3; ++m) {
    if (use_session && m == 2) {
      __syncthreads();
#pragma unroll
      for (int j = 0; j < 4; ++j) {
        int task = tid + j * 128;
        int r = task >> 5, c4 = task & 31;
        *(float4*)&s_z[2][r][c4 * 4] =
            *(const float4*)&session[(long)(base + r) * DIM + c4 * 4];
      }
    } else {
      __syncthreads();                   // protect s_nb / s_X reuse
#pragma unroll
      for (int j = 0; j < 3; ++j) {
        int e = tid + j * 128;
        if (e < 16 * KNB) {
          int r = e / KNB, k = e % KNB;
          s_nb[r][k] = nbs[m][(long)s_idx[r] * KNB + k];
        }
      }
      __syncthreads();
      // ---- neighbor mean -> X[:, 128:256] (bf16) --------------------------
#pragma unroll
      for (int j = 0; j < 4; ++j) {
        int task = tid + j * 128;
        int r = task >> 5, c4 = task & 31;
        float4 acc = {0.f, 0.f, 0.f, 0.f};
        for (int k = 0; k < KNB; ++k) {
          const float4 v = *(const float4*)&tabs[m][(long)s_nb[r][k] * DIM + c4 * 4];
          acc.x += v.x; acc.y += v.y; acc.z += v.z; acc.w += v.w;
        }
        const float inv = 1.0f / (float)KNB;
        v4bf p;
        p[0] = (__bf16)(acc.x * inv); p[1] = (__bf16)(acc.y * inv);
        p[2] = (__bf16)(acc.z * inv); p[3] = (__bf16)(acc.w * inv);
        *(v4bf*)&s_X[r][128 + c4 * 4] = p;
      }
      __syncthreads();
      // ---- C[16,128] = X[16,256] @ W  (bf16 WMMA, f32 acc) ----------------
      v8f c0 = {0.f, 0.f, 0.f, 0.f, 0.f, 0.f, 0.f, 0.f};
      v8f c1 = {0.f, 0.f, 0.f, 0.f, 0.f, 0.f, 0.f, 0.f};
#pragma unroll
      for (int ks = 0; ks < 8; ++ks) {
        const int k0 = ks * 32;
        const __bf16* ap = &s_X[l15][k0 + hf * 8];
        v16bf a = mk16(*(const v8bf*)ap, *(const v8bf*)(ap + 16));
        const __bf16* bpa = Wts[m] + (long)cola * 256 + k0 + hf * 8;
        v16bf ba = mk16(*(const v8bf*)bpa, *(const v8bf*)(bpa + 16));
        const __bf16* bpb = Wts[m] + (long)colb * 256 + k0 + hf * 8;
        v16bf bb = mk16(*(const v8bf*)bpb, *(const v8bf*)(bpb + 16));
        c0 = __builtin_amdgcn_wmma_f32_16x16x32_bf16(false, a, false, ba,
                                                     (short)0, c0, false, false);
        c1 = __builtin_amdgcn_wmma_f32_16x16x32_bf16(false, a, false, bb,
                                                     (short)0, c1, false, false);
      }
      const float bva = biases[m][cola];
      const float bvb = biases[m][colb];
#pragma unroll
      for (int g = 0; g < 8; ++g) {
        const int rowM = g + hf * 8;     // C layout: lanes>=16 hold M=8..15
        s_z[m][rowM][cola] = fmaxf(c0[g] + bva, 0.0f);
        s_z[m][rowM][colb] = fmaxf(c1[g] + bvb, 0.0f);
      }
    }
  }
  __syncthreads();

  // ---- semantic attention scoring: w[m][row] = sum_c tanh((z@Ws)+bs)*q ----
#pragma unroll
  for (int m = 0; m < 3; ++m) {
    v8f c0 = {0.f, 0.f, 0.f, 0.f, 0.f, 0.f, 0.f, 0.f};
    v8f c1 = {0.f, 0.f, 0.f, 0.f, 0.f, 0.f, 0.f, 0.f};
#pragma unroll
    for (int ks = 0; ks < 4; ++ks) {
      const int k0 = ks * 32;
      const float* ap = &s_z[m][l15][k0 + hf * 8];
      v16bf a = mk16(cvt8(ap), cvt8(ap + 16));
      const __bf16* bpa = Wst + (long)cola * 128 + k0 + hf * 8;
      v16bf ba = mk16(*(const v8bf*)bpa, *(const v8bf*)(bpa + 16));
      const __bf16* bpb = Wst + (long)colb * 128 + k0 + hf * 8;
      v16bf bb = mk16(*(const v8bf*)bpb, *(const v8bf*)(bpb + 16));
      c0 = __builtin_amdgcn_wmma_f32_16x16x32_bf16(false, a, false, ba,
                                                   (short)0, c0, false, false);
      c1 = __builtin_amdgcn_wmma_f32_16x16x32_bf16(false, a, false, bb,
                                                   (short)0, c1, false, false);
    }
    const float qa = q[cola], qb = q[colb];
    const float sa = bs[cola], sb = bs[colb];
#pragma unroll
    for (int g = 0; g < 8; ++g) {
      float s = tanhf(c0[g] + sa) * qa + tanhf(c1[g] + sb) * qb;
      s += __shfl_xor(s, 1);
      s += __shfl_xor(s, 2);
      s += __shfl_xor(s, 4);
      s += __shfl_xor(s, 8);             // reduce 16 lanes within each half
      if (l15 == 0) atomicAdd(&s_w[m][g + hf * 8], s);
    }
  }
  __syncthreads();

  if (tid < 3) {
    float t = 0.0f;
    for (int r = 0; r < 16; ++r) t += s_w[tid][r];
    atomicAdd(&wsum[tid], t);
  }

  // ---- write z tile to workspace: z_out[n][3][128] ------------------------
#pragma unroll
  for (int j = 0; j < 12; ++j) {
    int task = tid + j * 128;            // 3*16*32 float4
    int mm = task >> 9, rr = (task >> 5) & 15, c4 = task & 31;
    *(float4*)&z_out[((long)(base + rr) * 3 + mm) * DIM + c4 * 4] =
        *(const float4*)&s_z[mm][rr][c4 * 4];
  }
}

__global__ void softmax_beta_kernel(const float* __restrict__ wsum,
                                    float* __restrict__ beta) {
  int s = threadIdx.x;
  if (s < 3) {
    const float invN[3] = {1.0f / 51200.0f, 1.0f / 1024.0f, 1.0f / 1024.0f};
    float a0 = wsum[s * 3 + 0] * invN[s];
    float a1 = wsum[s * 3 + 1] * invN[s];
    float a2 = wsum[s * 3 + 2] * invN[s];
    float mx = fmaxf(a0, fmaxf(a1, a2));
    float e0 = expf(a0 - mx), e1 = expf(a1 - mx), e2 = expf(a2 - mx);
    float inv = 1.0f / (e0 + e1 + e2);
    beta[s * 3 + 0] = e0 * inv;
    beta[s * 3 + 1] = e1 * inv;
    beta[s * 3 + 2] = e2 * inv;
  }
}

__global__ void combine_kernel(const float* __restrict__ z,
                               const float* __restrict__ beta,
                               float* __restrict__ out, int N) {
  int t = blockIdx.x * blockDim.x + threadIdx.x;   // over N*32 float4
  if (t < N * 32) {
    int n = t >> 5, c4 = t & 31;
    const float b0 = beta[0], b1 = beta[1], b2 = beta[2];
    const long zb = (long)n * 3 * DIM + c4 * 4;
    const float4 z0 = *(const float4*)&z[zb];
    const float4 z1 = *(const float4*)&z[zb + DIM];
    const float4 z2 = *(const float4*)&z[zb + 2 * DIM];
    float4 o;
    o.x = b0 * z0.x + b1 * z1.x + b2 * z2.x;
    o.y = b0 * z0.y + b1 * z1.y + b2 * z2.y;
    o.z = b0 * z0.z + b1 * z1.z + b2 * z2.z;
    o.w = b0 * z0.w + b1 * z1.w + b2 * z2.w;
    *(float4*)&out[(long)n * DIM + c4 * 4] = o;
  }
}

extern "C" void kernel_launch(void* const* d_in, const int* in_sizes, int n_in,
                              void* d_out, int out_size, void* d_ws,
                              size_t ws_size, hipStream_t stream) {
  (void)in_sizes; (void)n_in; (void)out_size; (void)ws_size;

  const int*   item     = (const int*)d_in[0];
  const int*   locs     = (const int*)d_in[1];
  const int*   times    = (const int*)d_in[2];
  const float* session  = (const float*)d_in[3];
  const float* loc_emb  = (const float*)d_in[4];
  const float* time_emb = (const float*)d_in[5];
  const float* item_emb = (const float*)d_in[6];
  const int* nb_IL = (const int*)d_in[7];
  const int* nb_TL = (const int*)d_in[8];
  const int* nb_IT = (const int*)d_in[9];
  const int* nb_LT = (const int*)d_in[10];
  const int* nb_II = (const int*)d_in[11];
  const int* nb_LI = (const int*)d_in[12];
  const int* nb_TI = (const int*)d_in[13];
  const float* W_IL = (const float*)d_in[14]; const float* b_IL = (const float*)d_in[15];
  const float* W_TL = (const float*)d_in[16]; const float* b_TL = (const float*)d_in[17];
  const float* W_IT = (const float*)d_in[18]; const float* b_IT = (const float*)d_in[19];
  const float* W_LT = (const float*)d_in[20]; const float* b_LT = (const float*)d_in[21];
  const float* W_II = (const float*)d_in[22]; const float* b_II = (const float*)d_in[23];
  const float* W_LI = (const float*)d_in[24]; const float* b_LI = (const float*)d_in[25];
  const float* W_TI = (const float*)d_in[26]; const float* b_TI = (const float*)d_in[27];
  const float* Ws_l = (const float*)d_in[28]; const float* bs_l = (const float*)d_in[29];
  const float* q_l  = (const float*)d_in[30];
  const float* Ws_t = (const float*)d_in[31]; const float* bs_t = (const float*)d_in[32];
  const float* q_t  = (const float*)d_in[33];
  const float* Ws_i = (const float*)d_in[34]; const float* bs_i = (const float*)d_in[35];
  const float* q_i  = (const float*)d_in[36];

  const int NI_ROWS = 51200;   // B*L
  const int NB_ROWS = 1024;    // B

  // ---- workspace layout (~82.4 MB) ---------------------------------------
  float* z_i  = (float*)d_ws;                         // 51200*3*128
  float* z_l  = z_i + (size_t)NI_ROWS * 3 * DIM;      // 1024*3*128
  float* z_t  = z_l + (size_t)NB_ROWS * 3 * DIM;
  float* gws  = z_t + (size_t)NB_ROWS * 3 * DIM;      // 9 used, 16 padded
  float* beta = gws + 16;                             // 9 used, 16 padded
  __bf16* wt  = (__bf16*)(beta + 16);                 // 7 * 128*256 bf16
  __bf16* wst = wt + (size_t)7 * 32768;               // 3 * 128*128 bf16

  hipMemsetAsync(gws, 0, 16 * sizeof(float), stream);

  // ---- pass 1: weight transpose + bf16 convert ---------------------------
  // agg slots: [0]=II [1]=TI [2]=LI  (items, stack order II,TI,LI)
  //            [3]=IL [4]=TL        (locs)   [5]=IT [6]=LT (times)
  const float* aggW[7] = {W_II, W_TI, W_LI, W_IL, W_TL, W_IT, W_LT};
  for (int i = 0; i < 7; ++i)
    transpose_bf16_kernel<<<(32768 + 255) / 256, 256, 0, stream>>>(
        aggW[i], wt + (size_t)i * 32768, 256, 128);
  const float* semW[3] = {Ws_i, Ws_l, Ws_t};
  for (int i = 0; i < 3; ++i)
    transpose_bf16_kernel<<<(16384 + 255) / 256, 256, 0, stream>>>(
        semW[i], wst + (size_t)i * 16384, 128, 128);

  // ---- pass 2: per-stream tile kernels -----------------------------------
  hete_tile_kernel<<<NI_ROWS / 16, 128, 0, stream>>>(
      item, 0, item_emb,
      nb_II, item_emb, b_II, wt + (size_t)0 * 32768,
      nb_TI, time_emb, b_TI, wt + (size_t)1 * 32768,
      nb_LI, loc_emb,  b_LI, wt + (size_t)2 * 32768,
      nullptr, wst + (size_t)0 * 16384, bs_i, q_i, z_i, gws + 0);

  hete_tile_kernel<<<NB_ROWS / 16, 128, 0, stream>>>(
      locs, 1, loc_emb,
      nb_IL, item_emb, b_IL, wt + (size_t)3 * 32768,
      nb_TL, time_emb, b_TL, wt + (size_t)4 * 32768,
      nullptr, nullptr, nullptr, nullptr,
      session, wst + (size_t)1 * 16384, bs_l, q_l, z_l, gws + 3);

  hete_tile_kernel<<<NB_ROWS / 16, 128, 0, stream>>>(
      times, 1, time_emb,
      nb_IT, item_emb, b_IT, wt + (size_t)5 * 32768,
      nb_LT, loc_emb,  b_LT, wt + (size_t)6 * 32768,
      nullptr, nullptr, nullptr, nullptr,
      session, wst + (size_t)2 * 16384, bs_t, q_t, z_t, gws + 6);

  // ---- pass 3: beta = softmax(mean(w)) -----------------------------------
  softmax_beta_kernel<<<1, 4, 0, stream>>>(gws, beta);

  // ---- pass 4: weighted combine ------------------------------------------
  float* out = (float*)d_out;
  combine_kernel<<<(NI_ROWS * 32 + 255) / 256, 256, 0, stream>>>(
      z_i, beta + 0, out, NI_ROWS);
  combine_kernel<<<(NB_ROWS * 32 + 255) / 256, 256, 0, stream>>>(
      z_l, beta + 3, out + (size_t)NI_ROWS * DIM, NB_ROWS);
  combine_kernel<<<(NB_ROWS * 32 + 255) / 256, 256, 0, stream>>>(
      z_t, beta + 6, out + (size_t)NI_ROWS * DIM + (size_t)NB_ROWS * DIM, NB_ROWS);
}
